// LSTM_46548855554773
// MI455X (gfx1250) — compile-verified, matrix operand order checked by reading, not run on hardware
//
#include <hip/hip_runtime.h>
#include <hip/hip_bf16.h>

// ---------------------------------------------------------------------------
// 2-layer LSTM for MI455X (gfx1250, wave32, WMMA).
//   Phase 1: wx = x @ W^T + b  -> bf16 WMMA GEMM, double-buffered LDS tiles,
//            async global->LDS staging (GLOBAL_LOAD_ASYNC_TO_LDS_B128).
//   Phase 2: persistent recurrent kernel, U-slice resident in LDS (320KB/WGP),
//            h broadcast through L2 each step, atomic grid barrier.
// ---------------------------------------------------------------------------

typedef __bf16 v16bf __attribute__((ext_vector_type(16)));
typedef __bf16 v8bf  __attribute__((ext_vector_type(8)));
typedef __bf16 v4bf  __attribute__((ext_vector_type(4)));
typedef float  v8f   __attribute__((ext_vector_type(8)));
typedef int    v4i_  __attribute__((vector_size(16)));   // matches builtin sig

#define B_   32
#define T_   256
#define F_   512
#define H_   1024
#define G4_  4096          // 4*H
#define M_   8192          // B*T
#define APITCH 40          // GEMM LDS row pitch (halves), 80B, 16B aligned
#define UPITCH 1032        // recurrent LDS row pitch (halves): 2064B = 516 dw,
                           // 516 % 64 = 4 -> 16 lanes x b128 hit all 64 banks

// ---- async global->LDS (gfx1250 GLOBAL_LOAD_ASYNC_TO_LDS_B128) ------------
#if defined(__has_builtin)
#if __has_builtin(__builtin_amdgcn_global_load_async_to_lds_b128)
#define ASYNC_MODE 1
#endif
#endif
#ifndef ASYNC_MODE
#define ASYNC_MODE 0
#endif

typedef __attribute__((address_space(1))) v4i_ gv4i;   // global <4 x i32>
typedef __attribute__((address_space(3))) v4i_ lv4i;   // LDS    <4 x i32>

// copy 16B (8 bf16) global -> LDS
__device__ __forceinline__ void cp16(const __bf16* g, __bf16* l) {
#if ASYNC_MODE == 1
  __builtin_amdgcn_global_load_async_to_lds_b128((gv4i*)g, (lv4i*)l, 0, 0);
#else
  *(v8bf*)l = *(const v8bf*)g;
#endif
}

__device__ __forceinline__ void wait_async() {
#if ASYNC_MODE == 1
#if __has_builtin(__builtin_amdgcn_s_wait_asynccnt)
  __builtin_amdgcn_s_wait_asynccnt(0);
#endif
#endif
}

__device__ __forceinline__ v8f zero8() {
  v8f z;
#pragma unroll
  for (int j = 0; j < 8; ++j) z[j] = 0.f;
  return z;
}

// Load one 16x32 (A) or 32x16 (B) bf16 WMMA fragment from an LDS tile.
// lane l -> row (l&15), half-group (l>>4); halves j:
// k = 16*(j>=8) + 8*(l>>4) + (j&7)  (two contiguous 16B chunks).
__device__ __forceinline__ v16bf load_frag(const __bf16* tile, int pitch) {
  const int l = threadIdx.x & 31;
  const __bf16* p = tile + (l & 15) * pitch + (l >> 4) * 8;
  v8bf lo = *(const v8bf*)(p);
  v8bf hi = *(const v8bf*)(p + 16);
  v16bf r;
#pragma unroll
  for (int j = 0; j < 8; ++j) { r[j] = lo[j]; r[j + 8] = hi[j]; }
  return r;
}

__device__ __forceinline__ float sigmoid_f(float x) {
  return __builtin_amdgcn_rcpf(1.f + __expf(-x));
}
__device__ __forceinline__ float tanh_f(float x) {
  return 2.f * __builtin_amdgcn_rcpf(1.f + __expf(-2.f * x)) - 1.f;
}

// ---------------------------------------------------------------------------
// fp32 -> bf16 (vectorized x4)
// ---------------------------------------------------------------------------
__global__ void cvt_f32_bf16(const float* __restrict__ in,
                             __bf16* __restrict__ out, int n) {
  int i = (blockIdx.x * blockDim.x + threadIdx.x) * 4;
  if (i + 3 < n) {
    float4 v = *(const float4*)(in + i);
    v4bf o;
    o[0] = (__bf16)v.x; o[1] = (__bf16)v.y;
    o[2] = (__bf16)v.z; o[3] = (__bf16)v.w;
    *(v4bf*)(out + i) = o;
  }
}

// ---------------------------------------------------------------------------
// wx GEMM: C[m,n] = sum_k A[m,k]*W[n,k] + bias[n].  A:[M_,K] bf16, W:[4096,K]
// bf16 (row-major, K contiguous). Block tile 128M x 64N, 8 waves (4Mx2N),
// wave tile 32x32. Double-buffered LDS K-tiles: async-stage tile k+1 while
// WMMAs consume tile k.
// ---------------------------------------------------------------------------
__global__ __launch_bounds__(256)
void gemm_wx(const __bf16* __restrict__ A, const __bf16* __restrict__ W,
             const float* __restrict__ bias, float* __restrict__ Cout, int K) {
  __shared__ __align__(16) __bf16 As[2][128 * APITCH];
  __shared__ __align__(16) __bf16 Bs[2][64 * APITCH];
  const int tid = threadIdx.x;
  const int wid = tid >> 5;
  const int wm = wid >> 1, wn = wid & 1;
  const int mbase = blockIdx.y * 128;
  const int nbase = blockIdx.x * 64;

  v8f acc[2][2];
#pragma unroll
  for (int mi = 0; mi < 2; ++mi)
#pragma unroll
    for (int ni = 0; ni < 2; ++ni) acc[mi][ni] = zero8();

  const int arow = tid >> 1, aseg = tid & 1;   // A stage: 128 rows x 32 halves
  const int brow = tid >> 2, bseg = tid & 3;   // B stage: 64 rows x 32 halves
  const __bf16* Ag = A + (size_t)(mbase + arow) * K + aseg * 16;
  const __bf16* Bg = W + (size_t)(nbase + brow) * K + bseg * 8;
  __bf16* Al = &As[0][arow * APITCH + aseg * 16];
  __bf16* Bl = &Bs[0][brow * APITCH + bseg * 8];
  const int bufstepA = 128 * APITCH;   // halves between As[0] and As[1]
  const int bufstepB = 64 * APITCH;

  // prologue: stage tile 0 into buffer 0
  cp16(Ag, Al);
  cp16(Ag + 8, Al + 8);
  cp16(Bg, Bl);

  const int ntiles = K >> 5;
  for (int kt = 0; kt < ntiles; ++kt) {
    const int buf = kt & 1;
    wait_async();         // this wave's portion of tile kt landed in LDS
    __syncthreads();      // everyone's portion landed; prev compute finished
    if (kt + 1 < ntiles) {  // stage tile kt+1 into the other buffer (overlaps)
      const __bf16* ga = Ag + (kt + 1) * 32;
      const __bf16* gb = Bg + (kt + 1) * 32;
      __bf16* la = Al + (buf ^ 1) * bufstepA;
      __bf16* lb = Bl + (buf ^ 1) * bufstepB;
      cp16(ga, la);
      cp16(ga + 8, la + 8);
      cp16(gb, lb);
    }
    v16bf af[2], bfr[2];
    af[0]  = load_frag(&As[buf][(wm * 32 + 0 ) * APITCH], APITCH);
    af[1]  = load_frag(&As[buf][(wm * 32 + 16) * APITCH], APITCH);
    bfr[0] = load_frag(&Bs[buf][(wn * 32 + 0 ) * APITCH], APITCH);
    bfr[1] = load_frag(&Bs[buf][(wn * 32 + 16) * APITCH], APITCH);
#pragma unroll
    for (int mi = 0; mi < 2; ++mi)
#pragma unroll
      for (int ni = 0; ni < 2; ++ni)
        acc[mi][ni] = __builtin_amdgcn_wmma_f32_16x16x32_bf16(
            false, af[mi], false, bfr[ni], (short)0, acc[mi][ni], false, false);
  }

  const int l = tid & 31, nl = l & 15, hs = l >> 4;
#pragma unroll
  for (int ni = 0; ni < 2; ++ni) {
    const int n = nbase + wn * 32 + ni * 16 + nl;
    const float bv = bias[n];
#pragma unroll
    for (int mi = 0; mi < 2; ++mi)
#pragma unroll
      for (int r = 0; r < 8; ++r) {
        const int m = mbase + wm * 32 + mi * 16 + r + hs * 8;
        Cout[(size_t)m * G4_ + n] = acc[mi][ni][r] + bv;
      }
  }
}

// ---------------------------------------------------------------------------
// Persistent recurrent kernel. 64 blocks x 256 threads; block owns 16 hidden
// units j0..j0+15 -> 64 gate columns {j, H+j, 2H+j, 3H+j}. U-slice (64x1024
// bf16 = 128KB) lives in LDS for all 256 steps. Per step: async-stage h
// (32x1024 bf16) to LDS, issue wx[t] loads early (hide under WMMAs), 8 waves
// each produce one 16x16 D-fragment (32 WMMAs), exchange via LDS, fp32 gate
// math, ping-pong h state, atomic grid barrier.
// ---------------------------------------------------------------------------
__global__ __launch_bounds__(256)
void lstm_rec(const float* __restrict__ wx, const __bf16* __restrict__ Ubf,
              const float* __restrict__ ub,
              __bf16* __restrict__ hA, __bf16* __restrict__ hB,
              __bf16* __restrict__ seq_bf, float* __restrict__ seq_f32,
              float* __restrict__ hh, float* __restrict__ cc,
              unsigned* __restrict__ bar) {
  extern __shared__ __align__(16) char smem[];
  __bf16* u_lds = (__bf16*)smem;                                  // 64 x UPITCH
  __bf16* h_lds = (__bf16*)(smem + 64 * UPITCH * 2);              // 32 x UPITCH
  float*  gates = (float*)(smem + (64 + 32) * UPITCH * 2);        // 4*32*16 f32

  const int tid = threadIdx.x;
  const int wid = tid >> 5;
  const int l = tid & 31, nl = l & 15, hs = l >> 4;
  const int mt = wid & 1;          // batch half: rows 0-15 / 16-31
  const int nt = wid >> 1;         // gate index 0..3 (i,f,g,o)
  const int j0 = blockIdx.x * 16;
  const unsigned nblk = gridDim.x;

  // Preload U slice into LDS (resident for all timesteps).
  for (int i = tid; i < 64 * 128; i += 256) {
    const int c = i >> 7, ck = i & 127;
    const int g = (c >> 4) * H_ + j0 + (c & 15);
    cp16(Ubf + (size_t)g * H_ + ck * 8, &u_lds[c * UPITCH + ck * 8]);
  }

  // Per-thread pairs: p0=tid -> (b=tid>>4, j0+tid&15); p1=tid+256 -> b+16.
  const int b0 = tid >> 4, b1 = b0 + 16, jl = tid & 15;
  float creg[2] = {0.f, 0.f};
  const float ubias[4] = {ub[0 * H_ + j0 + jl], ub[1 * H_ + j0 + jl],
                          ub[2 * H_ + j0 + jl], ub[3 * H_ + j0 + jl]};
  wait_async();
  __syncthreads();

  for (int t = 0; t < T_; ++t) {
    const __bf16* hr = (t & 1) ? hB : hA;
    __bf16*       hw = (t & 1) ? hA : hB;

    // Stage full h [32 x 1024] into LDS (shared by the 4 N-waves per M-tile).
    for (int i = tid; i < 32 * 128; i += 256) {
      const int r = i >> 7, ck = i & 127;
      cp16(hr + r * H_ + ck * 8, &h_lds[r * UPITCH + ck * 8]);
    }
    // Issue this step's wx gate-bias loads now; latency hides under WMMAs.
    float wxv[2][4];
#pragma unroll
    for (int p = 0; p < 2; ++p) {
      const size_t wxrow = ((size_t)(p ? b1 : b0) * T_ + t) * G4_;
#pragma unroll
      for (int g = 0; g < 4; ++g)
        wxv[p][g] = wx[wxrow + g * H_ + j0 + jl];
    }
    wait_async();
    __syncthreads();

    v8f acc = zero8();
#pragma unroll 4
    for (int ks = 0; ks < 32; ++ks) {
      v16bf a = load_frag(&h_lds[(mt * 16) * UPITCH + ks * 32], UPITCH);
      v16bf b = load_frag(&u_lds[(nt * 16) * UPITCH + ks * 32], UPITCH);
      acc = __builtin_amdgcn_wmma_f32_16x16x32_bf16(
          false, a, false, b, (short)0, acc, false, false);
    }
#pragma unroll
    for (int r = 0; r < 8; ++r) {
      const int m = mt * 16 + r + hs * 8;            // batch row 0..31
      gates[nt * 512 + m * 16 + nl] = acc[r];
    }
    __syncthreads();

    if (t + 1 < T_)  // prefetch next step's wx row
      __builtin_prefetch(wx + ((size_t)b0 * T_ + t + 1) * G4_ + j0 + jl, 0, 0);

#pragma unroll
    for (int p = 0; p < 2; ++p) {
      const int bb = p ? b1 : b0;
      const float gi = gates[0 * 512 + bb * 16 + jl] + wxv[p][0] + ubias[0];
      const float gf = gates[1 * 512 + bb * 16 + jl] + wxv[p][1] + ubias[1];
      const float gg = gates[2 * 512 + bb * 16 + jl] + wxv[p][2] + ubias[2];
      const float go = gates[3 * 512 + bb * 16 + jl] + wxv[p][3] + ubias[3];
      const float iv = sigmoid_f(gi), fv = sigmoid_f(gf);
      const float gv = tanh_f(gg),    ov = sigmoid_f(go);
      const float c = fv * creg[p] + iv * gv;
      creg[p] = c;
      const float h = ov * tanh_f(c);
      const int j = j0 + jl;
      hw[bb * H_ + j] = (__bf16)h;
      if (seq_bf)  seq_bf[((size_t)bb * T_ + t) * H_ + j] = (__bf16)h;
      if (seq_f32) seq_f32[((size_t)bb * T_ + t) * H_ + j] = h;
      if (t == T_ - 1) { hh[bb * H_ + j] = h; cc[bb * H_ + j] = c; }
    }

    // Grid-wide barrier (monotonic counter, all 64 blocks resident).
    __syncthreads();
    if (tid == 0) {
      __threadfence();
      atomicAdd(bar, 1u);
      const unsigned target = nblk * (unsigned)(t + 1);
      while (atomicAdd(bar, 0u) < target) __builtin_amdgcn_s_sleep(1);
    }
    __syncthreads();
    __threadfence();
  }
}

// ---------------------------------------------------------------------------
extern "C" void kernel_launch(void* const* d_in, const int* in_sizes, int n_in,
                              void* d_out, int out_size, void* d_ws, size_t ws_size,
                              hipStream_t stream) {
  const float* x   = (const float*)d_in[0];
  const float* w0w = (const float*)d_in[1];
  const float* w0b = (const float*)d_in[2];
  const float* u0w = (const float*)d_in[3];
  const float* u0b = (const float*)d_in[4];
  const float* w1w = (const float*)d_in[5];
  const float* w1b = (const float*)d_in[6];
  const float* u1w = (const float*)d_in[7];
  const float* u1b = (const float*)d_in[8];

  float* out = (float*)d_out;                 // h2 [B,T,H]
  float* hh  = out + (size_t)M_ * H_;         // [2,B,H]
  float* cc  = hh + 2 * B_ * H_;              // [2,B,H]

  char* ws = (char*)d_ws;
  size_t off = 0;
  auto take = [&](size_t bytes) -> char* {
    char* p = ws + off;
    off += (bytes + 255) & ~(size_t)255;
    return p;
  };
  float*  wx   = (float*)take((size_t)M_ * G4_ * 4);     // reused for both layers
  __bf16* xbf  = (__bf16*)take((size_t)M_ * F_ * 2);
  __bf16* h1bf = (__bf16*)take((size_t)M_ * H_ * 2);     // layer-0 h sequence
  __bf16* w0bf = (__bf16*)take((size_t)G4_ * F_ * 2);
  __bf16* u0bf = (__bf16*)take((size_t)G4_ * H_ * 2);
  __bf16* w1bf = (__bf16*)take((size_t)G4_ * H_ * 2);
  __bf16* u1bf = (__bf16*)take((size_t)G4_ * H_ * 2);
  __bf16* hA   = (__bf16*)take((size_t)B_ * H_ * 2);
  __bf16* hB   = (__bf16*)take((size_t)B_ * H_ * 2);
  unsigned* bar0 = (unsigned*)take(256);
  unsigned* bar1 = (unsigned*)take(256);

  (void)hipMemsetAsync(hA, 0, (size_t)B_ * H_ * 2, stream);
  (void)hipMemsetAsync(hB, 0, (size_t)B_ * H_ * 2, stream);
  (void)hipMemsetAsync(bar0, 0, 256, stream);
  (void)hipMemsetAsync(bar1, 0, 256, stream);

  auto cvt = [&](const float* in, __bf16* o, int n) {
    cvt_f32_bf16<<<(n / 4 + 255) / 256, 256, 0, stream>>>(in, o, n);
  };
  cvt(x,   xbf,  M_ * F_);
  cvt(w0w, w0bf, G4_ * F_);
  cvt(u0w, u0bf, G4_ * H_);
  cvt(w1w, w1bf, G4_ * H_);
  cvt(u1w, u1bf, G4_ * H_);

  const int smem = (64 + 32) * UPITCH * 2 + 4 * 32 * 16 * 4;  // 206336 B
  (void)hipFuncSetAttribute(reinterpret_cast<const void*>(lstm_rec),
                            hipFuncAttributeMaxDynamicSharedMemorySize, smem);

  dim3 ggrid(G4_ / 64, M_ / 128);

  // Layer 0
  gemm_wx<<<ggrid, 256, 0, stream>>>(xbf, w0bf, w0b, wx, F_);
  lstm_rec<<<64, 256, smem, stream>>>(wx, u0bf, u0b, hA, hB,
                                      h1bf, nullptr, hh, cc, bar0);
  // Layer 1
  gemm_wx<<<ggrid, 256, 0, stream>>>(h1bf, w1bf, w1b, wx, H_);
  (void)hipMemsetAsync(hA, 0, (size_t)B_ * H_ * 2, stream);
  (void)hipMemsetAsync(hB, 0, (size_t)B_ * H_ * 2, stream);
  lstm_rec<<<64, 256, smem, stream>>>(wx, u1bf, u1b, hA, hB,
                                      nullptr, out, hh + B_ * H_, cc + B_ * H_, bar1);
}
